// RecurrentMemoryTransformer_76587856822353
// MI455X (gfx1250) — compile-verified, matrix operand order checked by reading, not run on hardware
//
#include <hip/hip_runtime.h>

// ---------------------------------------------------------------------------
// RecurrentMemoryTransformer forward for MI455X (gfx1250, wave32, WMMA).
//
// Algebraic refactor: NQ=18 << S=4096, so
//   scores = (q_h @ Wk_h) @ hs^T      (QK precomputed: 72x2048)
//   out    = ((attn @ hs) @ Wv_h^T)   (ctx = attn @ hs: 288x2048)
// which removes the 2 x 137 GFLOP K/V projections entirely. The op is then
// bound by two streaming passes over hidden_states (2 x 134 MB ~ 11.5 us at
// 23.3 TB/s). Both hs-touching GEMMs use v_wmma_f32_16x16x32_bf16; each wave
// carries 8 accumulator tiles (4 heads x 2 M-tiles) so every hs B-fragment is
// loaded from memory exactly once, and all 8 A-fragments are batch-loaded
// before the WMMA chain so loads overlap matrix issue.
// ---------------------------------------------------------------------------

#define H_    2048
#define K_    16
#define NQ_   18
#define NH_   4
#define HD_   512
#define FFN_  4096
#define MAXE_ 128
#define B_    4
#define S_    4096
#define EPS_  1e-6f
#define QPAD  32          // NQ padded to 2 WMMA M-tiles
#define INV_SQRT_HD 0.04419417382415922f  // 1/sqrt(512)

typedef __attribute__((ext_vector_type(16))) __bf16 v16bf;
typedef __attribute__((ext_vector_type(8)))  float  v8f;

// ---------------- workspace layout (bytes) ----------------
#define OFF_Q    0u                       // q_f32: 18*2048*4        = 147456
#define OFF_QKB  147456u                  // QKb bf16: 4*32*2048*2   = 524288
#define OFF_SC   671744u                  // scores: 16*32*4096*4    = 8388608
#define OFF_AT   9060352u                 // attn bf16: 16*32*4096*2 = 4194304
#define OFF_CTX  13254656u                // ctx: 16*32*2048*4       = 4194304
#define OFF_OUT  17448960u                // outp: 4*18*2048*4       = 589824
#define OFF_EX   18038784u                // extracted: 4*18*2048*4  = 589824
#define OFF_XN   18628608u                // xn: 64*2048*4           = 524288
#define OFF_GU   19152896u                // gate*up: 64*4096*4      = 1048576
#define OFF_H1   20201472u                // h1: 4*2048*4            = 32768
#define OFF_H2   20234240u                // h2: 4*512*4             = 8192
// total ~20.3 MB

__device__ __forceinline__ float sigmoidf_(float x) { return 1.f / (1.f + __expf(-x)); }
__device__ __forceinline__ float siluf_(float x)    { return x * sigmoidf_(x); }

// ---------------- WMMA fragment helpers (gfx1250 wave32 layouts) ----------
// A (16x32 bf16, row-major src): lane&15 = M; lanes<16 hold K {0..7,16..23},
// lanes>=16 hold K {8..15,24..31} (two contiguous 8-element runs).
__device__ __forceinline__ v16bf frag_a_bf16(const __bf16* A, int lda, int lane) {
  int m  = lane & 15;
  int kb = (lane < 16) ? 0 : 8;
  const __bf16* r = A + (long)m * lda + kb;
  v16bf f;
#pragma unroll
  for (int i = 0; i < 8; ++i) f[i] = r[i];
#pragma unroll
  for (int i = 0; i < 8; ++i) f[8 + i] = r[16 + i];
  return f;
}

// B (32x16 bf16) gathered from fp32 with strides sk (between K rows) and
// sn (between N cols); converts fp32 -> bf16 on the fly.
__device__ __forceinline__ v16bf frag_b_f32(const float* Bp, long sk, long sn, int lane) {
  int n  = lane & 15;
  int kb = (lane < 16) ? 0 : 8;
  const float* base = Bp + (long)n * sn + (long)kb * sk;
  v16bf f;
#pragma unroll
  for (int i = 0; i < 8; ++i) f[i] = (__bf16)base[(long)i * sk];
#pragma unroll
  for (int i = 0; i < 8; ++i) f[8 + i] = (__bf16)base[(long)(16 + i) * sk];
  return f;
}

// C/D 16x16 f32: VGPR r -> M = r (lanes 0-15) / M = 8+r (lanes 16-31), N = lane&15.
__device__ __forceinline__ void store_c(float* C, int ldc, int lane, v8f c) {
  int n  = lane & 15;
  int mb = (lane < 16) ? 0 : 8;
#pragma unroll
  for (int r = 0; r < 8; ++r) C[(long)(mb + r) * ldc + n] = c[r];
}

// ---------------- generic small GEMM (thread-per-output) ------------------
__global__ void matvec_kernel(const float* __restrict__ X, const float* __restrict__ W,
                              float* __restrict__ Y, int rows, int cols, int inner) {
  long idx = (long)blockIdx.x * blockDim.x + threadIdx.x;
  if (idx >= (long)rows * cols) return;
  int c = idx % cols;
  long r = idx / cols;
  const float* x = X + r * inner;
  const float* w = W + (long)c * inner;
  float a = 0.f;
#pragma unroll 4
  for (int i = 0; i < inner; ++i) a = fmaf(x[i], w[i], a);
  Y[idx] = a;
}

// QK[h][q, j] = (1/sqrt(HD)) * sum_d q[q, h*HD+d] * Wk[h*HD+d, j], padded to QPAD rows.
__global__ void qk_kernel(const float* __restrict__ q, const float* __restrict__ Wk,
                          __bf16* __restrict__ QKb) {
  long idx = (long)blockIdx.x * blockDim.x + threadIdx.x;
  if (idx >= (long)NH_ * QPAD * H_) return;
  int j   = idx % H_;
  int row = idx / H_;
  int h   = row / QPAD;
  int qi  = row % QPAD;
  float acc = 0.f;
  if (qi < NQ_) {
    const float* qp = q + (long)qi * H_ + (long)h * HD_;
    const float* wk = Wk + ((long)h * HD_) * H_ + j;
#pragma unroll 4
    for (int d = 0; d < HD_; ++d) acc = fmaf(qp[d], wk[(long)d * H_], acc);
    acc *= INV_SQRT_HD;
  }
  QKb[idx] = (__bf16)acc;
}

// scores: one wave per (s-tile, b); 8 accumulators = 4 heads x 2 M-tiles.
// hs B-fragment loaded once per K-step; all 8 A-fragments batch-loaded so the
// A-load clause overlaps the 8-WMMA chain.
__global__ void scores_wmma(const __bf16* __restrict__ QKb, const float* __restrict__ hs,
                            float* __restrict__ scores) {
  int lane = threadIdx.x;
  int st = blockIdx.x;          // S/16 tiles
  int b  = blockIdx.y;          // B
  const float* B0 = hs + ((long)b * S_ + (long)st * 16) * H_;   // B[k,n]=hs[s0+n, k]
  v8f c[8];
#pragma unroll
  for (int i = 0; i < 8; ++i) c[i] = (v8f){};
  for (int kt = 0; kt < H_ / 32; ++kt) {
    __builtin_prefetch(B0 + (kt + 4) * 32, 0, 1);               // global_prefetch_b8
    v16bf bb = frag_b_f32(B0 + kt * 32, /*sk=*/1, /*sn=*/H_, lane);
    v16bf a[8];
#pragma unroll
    for (int hm = 0; hm < 8; ++hm)                              // hm = h*2 + mt
      a[hm] = frag_a_bf16(QKb + ((long)hm * 16) * H_ + kt * 32, H_, lane);
#pragma unroll
    for (int hm = 0; hm < 8; ++hm)
      c[hm] = __builtin_amdgcn_wmma_f32_16x16x32_bf16(false, a[hm], false, bb,
                                                      (short)0, c[hm], false, false);
  }
#pragma unroll
  for (int hm = 0; hm < 8; ++hm) {
    float* C = scores + ((long)b * NH_ * QPAD + hm * 16) * S_ + (long)st * 16;
    store_c(C, S_, lane, c[hm]);
  }
}

// per-row softmax over S; writes bf16 attn (pad rows -> 0 so ctx GEMM ignores them)
__global__ void softmax_kernel(const float* __restrict__ scores, __bf16* __restrict__ attnb) {
  int row = blockIdx.x;                 // B*NH*QPAD rows
  int q   = row & (QPAD - 1);
  const float* src = scores + (long)row * S_;
  __bf16* dst = attnb + (long)row * S_;
  int t = threadIdx.x;
  __shared__ float red[256];
  if (q >= NQ_) {
    for (int i = t; i < S_; i += 256) dst[i] = (__bf16)0.f;
    return;
  }
  float m = -__builtin_inff();
  for (int i = t; i < S_; i += 256) m = fmaxf(m, src[i]);
  red[t] = m; __syncthreads();
  for (int o = 128; o; o >>= 1) { if (t < o) red[t] = fmaxf(red[t], red[t + o]); __syncthreads(); }
  m = red[0]; __syncthreads();
  float s = 0.f;
  for (int i = t; i < S_; i += 256) s += __expf(src[i] - m);
  red[t] = s; __syncthreads();
  for (int o = 128; o; o >>= 1) { if (t < o) red[t] += red[t + o]; __syncthreads(); }
  float inv = 1.f / red[0];
  for (int i = t; i < S_; i += 256) dst[i] = (__bf16)(__expf(src[i] - m) * inv);
}

// ctx: one wave per (n-tile, b); 8 accumulators = 4 heads x 2 M-tiles.
// hs B-fragment loaded once per K-step; A-fragments batch-loaded.
__global__ void ctx_wmma(const __bf16* __restrict__ attnb, const float* __restrict__ hs,
                         float* __restrict__ ctx) {
  int lane = threadIdx.x;
  int nt = blockIdx.x;          // H/16 tiles
  int b  = blockIdx.y;          // B
  const float* B0 = hs + (long)b * S_ * H_ + (long)nt * 16;     // B[k,n]=hs[k, n0+n]
  v8f c[8];
#pragma unroll
  for (int i = 0; i < 8; ++i) c[i] = (v8f){};
  for (int kt = 0; kt < S_ / 32; ++kt) {
    __builtin_prefetch(B0 + (long)(kt + 2) * 32 * H_, 0, 1);
    v16bf bb = frag_b_f32(B0 + (long)kt * 32 * H_, /*sk=*/H_, /*sn=*/1, lane);
    v16bf a[8];
#pragma unroll
    for (int hm = 0; hm < 8; ++hm)
      a[hm] = frag_a_bf16(attnb + ((long)b * NH_ * QPAD + hm * 16) * S_ + kt * 32,
                          S_, lane);
#pragma unroll
    for (int hm = 0; hm < 8; ++hm)
      c[hm] = __builtin_amdgcn_wmma_f32_16x16x32_bf16(false, a[hm], false, bb,
                                                      (short)0, c[hm], false, false);
  }
#pragma unroll
  for (int hm = 0; hm < 8; ++hm) {
    float* C = ctx + ((long)b * NH_ * QPAD + hm * 16) * H_ + (long)nt * 16;
    store_c(C, H_, lane, c[hm]);
  }
}

// out[b,q, h*HD+d] = ctx[b,h,q,:] . Wv[h*HD+d, :]
__global__ void outproj_kernel(const float* __restrict__ ctx, const float* __restrict__ Wv,
                               float* __restrict__ outp) {
  long idx = (long)blockIdx.x * blockDim.x + threadIdx.x;
  if (idx >= (long)B_ * NQ_ * H_) return;
  int  t  = idx % H_;
  long bq = idx / H_;
  int  b  = bq / NQ_;
  int  q  = bq % NQ_;
  int  h  = t / HD_;
  const float* x = ctx + (((long)(b * NH_ + h)) * QPAD + q) * H_;
  const float* w = Wv + (long)t * H_;
  float a = 0.f;
#pragma unroll 4
  for (int i = 0; i < H_; ++i) a = fmaf(x[i], w[i], a);
  outp[idx] = a;
}

// RMSNorm over the K memory rows
__global__ void rmsnorm_kernel(const float* __restrict__ extr, const float* __restrict__ ln_w,
                               float* __restrict__ xn) {
  int row = blockIdx.x;                 // 0..B*K-1
  int b = row / K_, k = row % K_;
  const float* x = extr + ((long)b * NQ_ + k) * H_;
  __shared__ float red[256];
  int t = threadIdx.x;
  float s = 0.f;
  for (int i = t; i < H_; i += 256) { float v = x[i]; s = fmaf(v, v, s); }
  red[t] = s; __syncthreads();
  for (int o = 128; o; o >>= 1) { if (t < o) red[t] += red[t + o]; __syncthreads(); }
  float inv = rsqrtf(red[0] / (float)H_ + EPS_);
  for (int i = t; i < H_; i += 256) xn[(long)row * H_ + i] = x[i] * inv * ln_w[i];
}

__global__ void gateup_kernel(const float* __restrict__ xn, const float* __restrict__ Wg,
                              const float* __restrict__ Wu, float* __restrict__ gu) {
  long idx = (long)blockIdx.x * blockDim.x + threadIdx.x;
  if (idx >= (long)B_ * K_ * FFN_) return;
  int f = idx % FFN_;
  int row = idx / FFN_;
  const float* x = xn + (long)row * H_;
  const float* g = Wg + (long)f * H_;
  const float* u = Wu + (long)f * H_;
  float ag = 0.f, au = 0.f;
#pragma unroll 4
  for (int i = 0; i < H_; ++i) { ag = fmaf(x[i], g[i], ag); au = fmaf(x[i], u[i], au); }
  gu[idx] = siluf_(ag) * au;
}

__global__ void downadd_kernel(const float* __restrict__ gu, const float* __restrict__ Wd,
                               const float* __restrict__ extr, float* __restrict__ mem_out) {
  long idx = (long)blockIdx.x * blockDim.x + threadIdx.x;
  if (idx >= (long)B_ * K_ * H_) return;
  int i = idx % H_;
  int row = idx / H_;
  int b = row / K_, k = row % K_;
  const float* gvec = gu + (long)row * FFN_;
  const float* w = Wd + (long)i * FFN_;
  float a = 0.f;
#pragma unroll 4
  for (int j = 0; j < FFN_; ++j) a = fmaf(gvec[j], w[j], a);
  mem_out[idx] = extr[((long)b * NQ_ + k) * H_ + i] + a;
}

// sigmoid(X[row] . W + bias) per row (used for priority and confidence)
__global__ void rowdot_sig_kernel(const float* __restrict__ X, long rstride,
                                  const float* __restrict__ W, const float* __restrict__ bias,
                                  float* __restrict__ out) {
  int row = blockIdx.x;
  const float* x = X + (long)row * rstride;
  __shared__ float red[256];
  int t = threadIdx.x;
  float s = 0.f;
  for (int i = t; i < H_; i += 256) s = fmaf(x[i], W[i], s);
  red[t] = s; __syncthreads();
  for (int o = 128; o; o >>= 1) { if (t < o) red[t] += red[t + o]; __syncthreads(); }
  if (t == 0) out[row] = sigmoidf_(red[0] + bias[0]);
}

// h1 = silu([value_vec ; hs[:, -1, :]] @ V1.T + b1)
__global__ void h1_kernel(const float* __restrict__ extr, const float* __restrict__ hs,
                          const float* __restrict__ V1, const float* __restrict__ b1,
                          float* __restrict__ h1) {
  long idx = (long)blockIdx.x * blockDim.x + threadIdx.x;
  if (idx >= (long)B_ * H_) return;
  int i = idx % H_;
  int b = idx / H_;
  const float* xv = extr + ((long)b * NQ_ + (K_ + 1)) * H_;
  const float* xh = hs + ((long)b * S_ + (S_ - 1)) * H_;
  const float* w = V1 + (long)i * (2 * H_);
  float a = 0.f;
#pragma unroll 4
  for (int j = 0; j < H_; ++j) a = fmaf(xv[j], w[j], a);
#pragma unroll 4
  for (int j = 0; j < H_; ++j) a = fmaf(xh[j], w[H_ + j], a);
  h1[idx] = siluf_(a + b1[i]);
}

__global__ void h2_kernel(const float* __restrict__ h1, const float* __restrict__ V2,
                          const float* __restrict__ b2, float* __restrict__ h2) {
  long idx = (long)blockIdx.x * blockDim.x + threadIdx.x;
  if (idx >= (long)B_ * (H_ / 4)) return;
  int i = idx % (H_ / 4);
  int b = idx / (H_ / 4);
  const float* x = h1 + (long)b * H_;
  const float* w = V2 + (long)i * H_;
  float a = 0.f;
#pragma unroll 4
  for (int j = 0; j < H_; ++j) a = fmaf(x[j], w[j], a);
  h2[idx] = siluf_(a + b2[i]);
}

__global__ void value_kernel(const float* __restrict__ h2, const float* __restrict__ V3,
                             const float* __restrict__ b3, float* __restrict__ out) {
  int b = blockIdx.x;
  __shared__ float red[128];
  int t = threadIdx.x;
  float s = 0.f;
  for (int i = t; i < H_ / 4; i += 128) s = fmaf(h2[(long)b * (H_ / 4) + i], V3[i], s);
  red[t] = s; __syncthreads();
  for (int o = 64; o; o >>= 1) { if (t < o) red[t] += red[t + o]; __syncthreads(); }
  if (t == 0) out[b] = red[0] + b3[0];
}

__global__ void copy_kernel(const float* __restrict__ src, float* __restrict__ dst, long n) {
  long idx = (long)blockIdx.x * blockDim.x + threadIdx.x;
  if (idx < n) dst[idx] = src[idx];
}

// sequential 16-step priority-buffer scan (single block, matches lax.scan)
__global__ void buffer_scan(const float* __restrict__ memory, const float* __restrict__ priority,
                            float* __restrict__ entries, float* __restrict__ pris) {
  __shared__ float bestv[128];
  __shared__ int   besti[128];
  int t = threadIdx.x;                  // blockDim == 128 == MAXE_
  for (int step = 0; step < K_; ++step) {
    bestv[t] = pris[t];
    besti[t] = t;
    __syncthreads();
    for (int o = 64; o; o >>= 1) {
      if (t < o) {
        bool sw = (bestv[t + o] < bestv[t]) ||
                  (bestv[t + o] == bestv[t] && besti[t + o] < besti[t]);
        if (sw) { bestv[t] = bestv[t + o]; besti[t] = besti[t + o]; }
      }
      __syncthreads();
    }
    int   idx  = besti[0];
    float minv = bestv[0];
    float p    = priority[step];        // priority[0, step]
    if (p > minv) {                     // uniform branch
      for (int i = t; i < H_; i += 128)
        entries[(long)idx * H_ + i] = memory[(long)step * H_ + i];
      if (t == 0) pris[idx] = p;
    }
    __syncthreads();
  }
}

extern "C" void kernel_launch(void* const* d_in, const int* in_sizes, int n_in,
                              void* d_out, int out_size, void* d_ws, size_t ws_size,
                              hipStream_t stream) {
  const float* hs      = (const float*)d_in[0];
  const float* bufE    = (const float*)d_in[1];
  const float* bufP    = (const float*)d_in[2];
  const float* queries = (const float*)d_in[3];
  const float* Wq = (const float*)d_in[4];
  const float* Wk = (const float*)d_in[5];
  const float* Wv = (const float*)d_in[6];
  const float* Wo = (const float*)d_in[7];
  const float* ln_w = (const float*)d_in[8];
  const float* Wg = (const float*)d_in[9];
  const float* Wu = (const float*)d_in[10];
  const float* Wd = (const float*)d_in[11];
  const float* Wp = (const float*)d_in[12];
  const float* bp = (const float*)d_in[13];
  const float* Wh = (const float*)d_in[14];
  const float* bh = (const float*)d_in[15];
  const float* V1 = (const float*)d_in[16];
  const float* b1 = (const float*)d_in[17];
  const float* V2 = (const float*)d_in[18];
  const float* b2 = (const float*)d_in[19];
  const float* V3 = (const float*)d_in[20];
  const float* b3 = (const float*)d_in[21];

  char* ws = (char*)d_ws;
  float*  q_f32  = (float*)(ws + OFF_Q);
  __bf16* QKb    = (__bf16*)(ws + OFF_QKB);
  float*  scores = (float*)(ws + OFF_SC);
  __bf16* attnb  = (__bf16*)(ws + OFF_AT);
  float*  ctx    = (float*)(ws + OFF_CTX);
  float*  outp   = (float*)(ws + OFF_OUT);
  float*  extr   = (float*)(ws + OFF_EX);
  float*  xn     = (float*)(ws + OFF_XN);
  float*  gu     = (float*)(ws + OFF_GU);
  float*  h1     = (float*)(ws + OFF_H1);
  float*  h2     = (float*)(ws + OFF_H2);

  float* o       = (float*)d_out;
  float* o_mem   = o;                               // (B,K,H)
  float* o_pri   = o_mem + (long)B_ * K_ * H_;      // (B,K)
  float* o_conf  = o_pri + B_ * K_;                 // (B,)
  float* o_val   = o_conf + B_;                     // (B,)
  float* o_ent   = o_val + B_;                      // (MAXE,H)
  float* o_bpri  = o_ent + (long)MAXE_ * H_;        // (MAXE,)

  // 1) q = queries @ Wq.T
  matvec_kernel<<<(NQ_ * H_ + 255) / 256, 256, 0, stream>>>(queries, Wq, q_f32, NQ_, H_, H_);
  // 2) QK[h] = (q_h @ Wk_h) / sqrt(HD), bf16, padded rows zero
  qk_kernel<<<((long)NH_ * QPAD * H_ + 255) / 256, 256, 0, stream>>>(q_f32, Wk, QKb);
  // 3) scores (WMMA bf16, first streaming pass over hs; hs read exactly once)
  scores_wmma<<<dim3(S_ / 16, B_), 32, 0, stream>>>(QKb, hs, scores);
  // 4) softmax -> bf16 attn
  softmax_kernel<<<B_ * NH_ * QPAD, 256, 0, stream>>>(scores, attnb);
  // 5) ctx = attn @ hs (WMMA bf16, second streaming pass; hs read exactly once)
  ctx_wmma<<<dim3(H_ / 16, B_), 32, 0, stream>>>(attnb, hs, ctx);
  // 6) out = ctx @ Wv_h.T ; extracted = out @ Wo.T
  outproj_kernel<<<((long)B_ * NQ_ * H_ + 255) / 256, 256, 0, stream>>>(ctx, Wv, outp);
  matvec_kernel<<<(B_ * NQ_ * H_ + 255) / 256, 256, 0, stream>>>(outp, Wo, extr, B_ * NQ_, H_, H_);
  // 7) RMSNorm + SwiGLU FFN + residual -> memory output
  rmsnorm_kernel<<<B_ * K_, 256, 0, stream>>>(extr, ln_w, xn);
  gateup_kernel<<<((long)B_ * K_ * FFN_ + 255) / 256, 256, 0, stream>>>(xn, Wg, Wu, gu);
  downadd_kernel<<<((long)B_ * K_ * H_ + 255) / 256, 256, 0, stream>>>(gu, Wd, extr, o_mem);
  // 8) priority / confidence heads
  rowdot_sig_kernel<<<B_ * K_, 256, 0, stream>>>(o_mem, (long)H_, Wp, bp, o_pri);
  rowdot_sig_kernel<<<B_, 256, 0, stream>>>(extr + (long)K_ * H_, (long)NQ_ * H_, Wh, bh, o_conf);
  // 9) value head
  h1_kernel<<<(B_ * H_ + 255) / 256, 256, 0, stream>>>(extr, hs, V1, b1, h1);
  h2_kernel<<<(B_ * (H_ / 4) + 255) / 256, 256, 0, stream>>>(h1, V2, b2, h2);
  value_kernel<<<B_, 128, 0, stream>>>(h2, V3, b3, o_val);
  // 10) buffer copy + sequential priority scan
  copy_kernel<<<((long)MAXE_ * H_ + 255) / 256, 256, 0, stream>>>(bufE, o_ent, (long)MAXE_ * H_);
  copy_kernel<<<1, 128, 0, stream>>>(bufP, o_bpri, MAXE_);
  buffer_scan<<<1, 128, 0, stream>>>(o_mem, o_pri, o_ent, o_bpri);
}
